// attention_module_multi_head_44985487459127
// MI455X (gfx1250) — matrix operand order repaired and checked by reading, not compiled, not run on hardware
//
#include <hip/hip_runtime.h>
#include <hip/hip_bf16.h>

typedef __attribute__((ext_vector_type(16))) __bf16 v16bf;
typedef __attribute__((ext_vector_type(8)))  __bf16 v8bf;
typedef __attribute__((ext_vector_type(8)))  float  v8f;
typedef __attribute__((ext_vector_type(4)))  unsigned int u32x4;
typedef __attribute__((ext_vector_type(8)))  int i32x8;
typedef __attribute__((ext_vector_type(4)))  int i32x4;

#define NROI   2048
#define NONGT  1024
#define GRP    16
static const long NM = (long)NROI * NONGT;   // rows*cols of one attention map

__device__ inline v8f zero8() {
  v8f v;
  #pragma unroll
  for (int i = 0; i < 8; ++i) v[i] = 0.f;
  return v;
}

__device__ inline __bf16 f2bf(float f) { return (__bf16)f; }

// ---- Tensor Data Mover: 2D bf16 tile -> LDS (D# per ISA 8.3-8.5) ----------
// Loads tile_rows x tile_cols (elements) from row-major global (row stride
// stride_elems) into LDS at lds_off, rows compacted (tile_cols contiguous).
__device__ inline void tdm_load_2d(unsigned lds_off, const void* gptr,
                                   unsigned tile_cols, unsigned tile_rows,
                                   unsigned stride_elems, unsigned tensor_rows) {
  unsigned long long ga = (unsigned long long)gptr;
  u32x4 g0;
  g0[0] = 1u;                                        // count=1, user descriptor
  g0[1] = lds_off;                                   // lds_addr (bytes)
  g0[2] = (unsigned)ga;                              // global_addr[31:0]
  g0[3] = (unsigned)((ga >> 32) & 0x01ffffffull)     // global_addr[56:32]
        | (2u << 30);                                // type=2 ("image")
  i32x8 g1;
  g1[0] = (int)(1u << 16);                           // data_size=1 -> 2 bytes
  g1[1] = (int)((stride_elems & 0xffffu) << 16);     // tensor_dim0[15:0]
  g1[2] = (int)((stride_elems >> 16)                 // tensor_dim0[31:16]
        | ((tensor_rows & 0xffffu) << 16));          // tensor_dim1[15:0]
  g1[3] = (int)((tensor_rows >> 16)                  // tensor_dim1[31:16]
        | (tile_cols << 16));                        // tile_dim0
  g1[4] = (int)(tile_rows & 0xffffu);                // tile_dim1 (tile_dim2=0)
  g1[5] = (int)stride_elems;                         // tensor_dim0_stride[31:0]
  g1[6] = 0;                                         // stride0[47:32]|stride1 lo
  g1[7] = 0;
  i32x4 z4; z4[0] = 0; z4[1] = 0; z4[2] = 0; z4[3] = 0;
  i32x8 z8;
  #pragma unroll
  for (int i = 0; i < 8; ++i) z8[i] = 0;
  __builtin_amdgcn_tensor_load_to_lds(g0, g1, z4, z4, z8, 0);
}

// ---- WMMA fragment loaders (bf16, 16x16x32) -------------------------------
// A matrix 16x32: lane0-15 -> M=lane, K=0..15 contiguous; lane16-31 -> K=16..31
__device__ inline v16bf load_frag_a(const __bf16* __restrict__ base, int row0,
                                    int ld, int k0) {
  int lane = threadIdx.x & 31;
  const __bf16* p = base + (long)(row0 + (lane & 15)) * ld + k0 + ((lane >> 4) << 4);
  v16bf f;
  #pragma unroll
  for (int e = 0; e < 16; ++e) f[e] = p[e];
  return f;
}
// B matrix 32x16 from a row-major [N][K] tile (A @ B^T form):
// lane0-15 -> N=lane, K=0..15 contiguous; lane16-31 -> K=16..31
__device__ inline v16bf load_frag_b_nt(const __bf16* __restrict__ base, int col0,
                                       int ld, int k0) {
  int lane = threadIdx.x & 31;
  const __bf16* p = base + (long)(col0 + (lane & 15)) * ld + k0 + ((lane >> 4) << 4);
  v16bf f;
  #pragma unroll
  for (int e = 0; e < 16; ++e) f[e] = p[e];
  return f;
}
// B matrix 32x16 from a row-major [K][N] tile (A @ B form)
__device__ inline v16bf load_frag_b_nn(const __bf16* __restrict__ base, int col0,
                                       int ldn, int k0) {
  int lane = threadIdx.x & 31;
  int c  = col0 + (lane & 15);
  int kb = k0 + ((lane >> 4) << 4);
  v16bf f;
  #pragma unroll
  for (int e = 0; e < 16; ++e) f[e] = base[(kb + e) * ldn + c];
  return f;
}
__device__ inline v8f wmma_bf16(v16bf a, v16bf b, v8f c) {
  return __builtin_amdgcn_wmma_f32_16x16x32_bf16(false, a, false, b, (short)0, c,
                                                 false, false);
}

// ---- K0: f32 -> bf16 convert ----------------------------------------------
__global__ __launch_bounds__(256) void cvt_kernel(const float* __restrict__ in,
                                                  __bf16* __restrict__ out, long n) {
  long i = (long)blockIdx.x * 256 + threadIdx.x;
  if (i < n) out[i] = f2bf(in[i]);
}

// ---- K1: C_bf16[M,N] = scale*(A_bf @ B_bf^T + bias) ------------------------
// 64x64 tile / workgroup (8 waves, each 16x32). TDM double-buffered staging.
__global__ __launch_bounds__(256) void gemm_nt_bf16(
    const __bf16* __restrict__ A, const __bf16* __restrict__ B,
    const float* __restrict__ bias, __bf16* __restrict__ C,
    int M, int N, int K, float scale) {
  __shared__ __bf16 lA[2][64 * 32];
  __shared__ __bf16 lB[2][64 * 32];
  int tid = threadIdx.x, lane = tid & 31, wv = tid >> 5;
  int m0 = blockIdx.y * 64, n0 = blockIdx.x * 64;
  int wm = (wv >> 1) * 16, wn = (wv & 1) * 32;
  v8f acc0 = zero8(), acc1 = zero8();
  if (wv == 0) {   // TENSORcnt is per-wave; wave0 drives the DMA
    tdm_load_2d((unsigned)(unsigned long long)lA[0], A + (long)m0 * K, 32, 64, K, M);
    tdm_load_2d((unsigned)(unsigned long long)lB[0], B + (long)n0 * K, 32, 64, K, N);
  }
  for (int k0 = 0; k0 < K; k0 += 32) {
    int cur = (k0 >> 5) & 1;
    if (wv == 0) {
      if (k0 + 32 < K) {
        tdm_load_2d((unsigned)(unsigned long long)lA[cur ^ 1],
                    A + (long)m0 * K + k0 + 32, 32, 64, K, M);
        tdm_load_2d((unsigned)(unsigned long long)lB[cur ^ 1],
                    B + (long)n0 * K + k0 + 32, 32, 64, K, N);
        __builtin_amdgcn_s_wait_tensorcnt(2);  // in-order: current tiles done
      } else {
        __builtin_amdgcn_s_wait_tensorcnt(0);
      }
    }
    __syncthreads();
    v16bf a  = load_frag_a(lA[cur], wm, 32, 0);
    v16bf b0 = load_frag_b_nt(lB[cur], wn, 32, 0);
    v16bf b1 = load_frag_b_nt(lB[cur], wn + 16, 32, 0);
    acc0 = wmma_bf16(a, b0, acc0);
    acc1 = wmma_bf16(a, b1, acc1);
    __syncthreads();
  }
  int c0 = n0 + wn + (lane & 15), c1 = c0 + 16;
  int r0 = m0 + wm + ((lane >> 4) << 3);
  float bv0 = bias[c0], bv1 = bias[c1];
  #pragma unroll
  for (int r = 0; r < 8; ++r) {
    C[(long)(r0 + r) * N + c0] = f2bf((acc0[r] + bv0) * scale);
    C[(long)(r0 + r) * N + c1] = f2bf((acc1[r] + bv1) * scale);
  }
}

// ---- K2: waff[g][n][m] = log(max(relu(PE[n,m,:] . Wpos[g,:] + bpos[g]), eps))
// wave handles 16 flat rows (n*1024+m); 8 waves/block -> 128 rows
__global__ __launch_bounds__(256) void posfeat_kernel(
    const float* __restrict__ pe, const __bf16* __restrict__ Wpos,
    const float* __restrict__ bpos, float* __restrict__ waff) {
  __shared__ __bf16 lA[8][16 * 64];
  int lane = threadIdx.x & 31, wv = threadIdx.x >> 5;
  long rbase = (long)blockIdx.x * 128 + wv * 16;
  const float* gp = pe + rbase * 64;
  __bf16* la = lA[wv];
  #pragma unroll
  for (int i = lane; i < 256; i += 32) {   // 16x64 f32 tile -> bf16 LDS
    float4 v = ((const float4*)gp)[i];
    la[i * 4 + 0] = f2bf(v.x); la[i * 4 + 1] = f2bf(v.y);
    la[i * 4 + 2] = f2bf(v.z); la[i * 4 + 3] = f2bf(v.w);
  }
  __builtin_prefetch(gp + 16 * 64, 0, 1);  // next PE tile into L2
  // wave-private tile: no workgroup barrier needed
  v16bf a0 = load_frag_a(la, 0, 64, 0);
  v16bf a1 = load_frag_a(la, 0, 64, 32);
  v16bf b0 = load_frag_b_nt(Wpos, 0, 64, 0);   // Wpos is (16,64) row-major
  v16bf b1 = load_frag_b_nt(Wpos, 0, 64, 32);
  v8f acc = zero8();
  acc = wmma_bf16(a0, b0, acc);
  acc = wmma_bf16(a1, b1, acc);
  int g = lane & 15;
  long n = rbase / 1024;
  int  mt = (int)(rbase & 1023) + ((lane >> 4) << 3);
  float bg = bpos[g];
  float* op = waff + (long)g * NM + n * 1024 + mt;
  #pragma unroll
  for (int r = 0; r < 8; ++r) {
    float v = fmaxf(acc[r] + bg, 0.f);        // relu
    op[r] = __logf(fmaxf(v, 1e-6f));          // log(max(.,eps))
  }
}

// ---- K3: waff += qb . kb^T  (scale folded into q); K=64, C-operand = waff --
__global__ __launch_bounds__(256) void affscale_kernel(
    const __bf16* __restrict__ Q, const __bf16* __restrict__ Kb,
    float* __restrict__ waff) {
  int lane = threadIdx.x & 31, wv = threadIdx.x >> 5;
  int g  = blockIdx.z;
  int n0 = blockIdx.y * 128 + wv * 16;
  int m0 = blockIdx.x * 32;
  int kq = g * 64;
  v16bf a0  = load_frag_a(Q, n0, 1024, kq);
  v16bf a1  = load_frag_a(Q, n0, 1024, kq + 32);
  v16bf b00 = load_frag_b_nt(Kb, m0, 1024, kq);
  v16bf b01 = load_frag_b_nt(Kb, m0, 1024, kq + 32);
  v16bf b10 = load_frag_b_nt(Kb, m0 + 16, 1024, kq);
  v16bf b11 = load_frag_b_nt(Kb, m0 + 16, 1024, kq + 32);
  float* base = waff + (long)g * NM;
  int col = m0 + (lane & 15);
  int r0  = n0 + ((lane >> 4) << 3);
  v8f c0, c1;
  #pragma unroll
  for (int r = 0; r < 8; ++r) {
    c0[r] = base[(long)(r0 + r) * 1024 + col];
    c1[r] = base[(long)(r0 + r) * 1024 + col + 16];
  }
  c0 = wmma_bf16(a0, b00, c0); c0 = wmma_bf16(a1, b01, c0);
  c1 = wmma_bf16(a0, b10, c1); c1 = wmma_bf16(a1, b11, c1);
  #pragma unroll
  for (int r = 0; r < 8; ++r) {
    base[(long)(r0 + r) * 1024 + col]      = c0[r];
    base[(long)(r0 + r) * 1024 + col + 16] = c1[r];
  }
}

// ---- K4: softmax over m (1024) per (g,n) row; one wave32 per row -----------
__global__ __launch_bounds__(256) void softmax_kernel(
    const float* __restrict__ waff, __bf16* __restrict__ P) {
  int lane = threadIdx.x & 31, wv = threadIdx.x >> 5;
  long row = (long)blockIdx.x * 8 + wv;
  const float* p = waff + row * 1024;
  float v[32], mx = -3.4e38f;
  #pragma unroll
  for (int i = 0; i < 32; ++i) { v[i] = p[lane + i * 32]; mx = fmaxf(mx, v[i]); }
  #pragma unroll
  for (int o = 16; o; o >>= 1) mx = fmaxf(mx, __shfl_xor(mx, o, 32));
  float s = 0.f;
  #pragma unroll
  for (int i = 0; i < 32; ++i) { v[i] = __expf(v[i] - mx); s += v[i]; }
  #pragma unroll
  for (int o = 16; o; o >>= 1) s += __shfl_xor(s, o, 32);
  float inv = 1.f / s;
  __bf16* q = P + row * 1024;
  #pragma unroll
  for (int i = 0; i < 32; ++i) q[lane + i * 32] = f2bf(v[i] * inv);
}

// ---- K5: OT[g] = P[g](2048x1024) @ V(1024x1024)  (NN form) -----------------
// 128x64 block tile, 8 waves each 32x32 (4 WMMA/k-step), TDM double-buffered.
__global__ __launch_bounds__(256) void outfeat_kernel(
    const __bf16* __restrict__ P, const __bf16* __restrict__ V,
    __bf16* __restrict__ OT) {
  __shared__ __bf16 lA[2][128 * 32];
  __shared__ __bf16 lB[2][32 * 64];
  int tid = threadIdx.x, lane = tid & 31, wv = tid >> 5;
  int g = blockIdx.z, n0 = blockIdx.y * 128, f0 = blockIdx.x * 64;
  const __bf16* A = P + (long)g * NM;
  int wm = (wv >> 1) * 32, wn = (wv & 1) * 32;
  v8f acc00 = zero8(), acc01 = zero8(), acc10 = zero8(), acc11 = zero8();
  if (wv == 0) {
    tdm_load_2d((unsigned)(unsigned long long)lA[0], A + (long)n0 * 1024,
                32, 128, 1024, NROI);
    tdm_load_2d((unsigned)(unsigned long long)lB[0], V + f0,
                64, 32, 1024, NONGT);
  }
  for (int k0 = 0; k0 < 1024; k0 += 32) {
    int cur = (k0 >> 5) & 1;
    if (wv == 0) {
      if (k0 + 32 < 1024) {
        tdm_load_2d((unsigned)(unsigned long long)lA[cur ^ 1],
                    A + (long)n0 * 1024 + k0 + 32, 32, 128, 1024, NROI);
        tdm_load_2d((unsigned)(unsigned long long)lB[cur ^ 1],
                    V + (long)(k0 + 32) * 1024 + f0, 64, 32, 1024, NONGT);
        __builtin_amdgcn_s_wait_tensorcnt(2);
      } else {
        __builtin_amdgcn_s_wait_tensorcnt(0);
      }
    }
    __syncthreads();
    v16bf a0 = load_frag_a(lA[cur], wm, 32, 0);
    v16bf a1 = load_frag_a(lA[cur], wm + 16, 32, 0);
    v16bf b0 = load_frag_b_nn(lB[cur], wn, 64, 0);
    v16bf b1 = load_frag_b_nn(lB[cur], wn + 16, 64, 0);
    acc00 = wmma_bf16(a0, b0, acc00);
    acc01 = wmma_bf16(a0, b1, acc01);
    acc10 = wmma_bf16(a1, b0, acc10);
    acc11 = wmma_bf16(a1, b1, acc11);
    __syncthreads();
  }
  __bf16* O = OT + (long)g * NM;
  int c0 = f0 + wn + (lane & 15), c1 = c0 + 16;
  int r0 = n0 + wm + ((lane >> 4) << 3);
  #pragma unroll
  for (int r = 0; r < 8; ++r) {
    O[(long)(r0 + r) * 1024 + c0]      = f2bf(acc00[r]);
    O[(long)(r0 + r) * 1024 + c1]      = f2bf(acc01[r]);
    O[(long)(r0 + r + 16) * 1024 + c0] = f2bf(acc10[r]);
    O[(long)(r0 + r + 16) * 1024 + c1] = f2bf(acc11[r]);
  }
}

// ---- K6: out[n][g*64+o] = OT[g][n][:] . Wc[g*64+o][:] + bconv --------------
__global__ __launch_bounds__(256) void conv_kernel(
    const __bf16* __restrict__ OT, const __bf16* __restrict__ Wc,
    const float* __restrict__ bconv, float* __restrict__ out) {
  __shared__ __bf16 lA[2][64 * 32];
  __shared__ __bf16 lB[2][64 * 32];
  int tid = threadIdx.x, lane = tid & 31, wv = tid >> 5;
  int g = blockIdx.z, n0 = blockIdx.y * 64;
  const __bf16* A = OT + (long)g * NM;
  const __bf16* B = Wc + (long)g * 64 * 1024;   // 64 rows (o) x 1024 (f)
  int wm = (wv >> 1) * 16, wn = (wv & 1) * 32;
  v8f acc0 = zero8(), acc1 = zero8();
  if (wv == 0) {
    tdm_load_2d((unsigned)(unsigned long long)lA[0], A + (long)n0 * 1024,
                32, 64, 1024, NROI);
    tdm_load_2d((unsigned)(unsigned long long)lB[0], B, 32, 64, 1024, 64);
  }
  for (int k0 = 0; k0 < 1024; k0 += 32) {
    int cur = (k0 >> 5) & 1;
    if (wv == 0) {
      if (k0 + 32 < 1024) {
        tdm_load_2d((unsigned)(unsigned long long)lA[cur ^ 1],
                    A + (long)n0 * 1024 + k0 + 32, 32, 64, 1024, NROI);
        tdm_load_2d((unsigned)(unsigned long long)lB[cur ^ 1],
                    B + k0 + 32, 32, 64, 1024, 64);
        __builtin_amdgcn_s_wait_tensorcnt(2);
      } else {
        __builtin_amdgcn_s_wait_tensorcnt(0);
      }
    }
    __syncthreads();
    v16bf a  = load_frag_a(lA[cur], wm, 32, 0);
    v16bf b0 = load_frag_b_nt(lB[cur], wn, 32, 0);
    v16bf b1 = load_frag_b_nt(lB[cur], wn + 16, 32, 0);
    acc0 = wmma_bf16(a, b0, acc0);
    acc1 = wmma_bf16(a, b1, acc1);
    __syncthreads();
  }
  int c0 = g * 64 + wn + (lane & 15), c1 = c0 + 16;
  int r0 = n0 + wm + ((lane >> 4) << 3);
  float bv0 = bconv[c0], bv1 = bconv[c1];
  #pragma unroll
  for (int r = 0; r < 8; ++r) {
    out[(long)(r0 + r) * 1024 + c0] = acc0[r] + bv0;
    out[(long)(r0 + r) * 1024 + c1] = acc1[r] + bv1;
  }
}

extern "C" void kernel_launch(void* const* d_in, const int* in_sizes, int n_in,
                              void* d_out, int out_size, void* d_ws, size_t ws_size,
                              hipStream_t stream) {
  const float* roi   = (const float*)d_in[0];   // (2048,1024)
  const float* pe    = (const float*)d_in[1];   // (2048,1024,64)
  const float* Wpos  = (const float*)d_in[2];   // (16,64)
  const float* bpos  = (const float*)d_in[3];
  const float* Wq    = (const float*)d_in[4];   // (1024,1024)
  const float* bq    = (const float*)d_in[5];
  const float* Wk    = (const float*)d_in[6];
  const float* bk    = (const float*)d_in[7];
  const float* Wconv = (const float*)d_in[8];   // (1024,1024)
  const float* bconv = (const float*)d_in[9];

  char* w = (char*)d_ws; size_t off = 0;
  auto alloc = [&](size_t bytes) -> void* {
    void* p = w + off; off += (bytes + 255) & ~(size_t)255; return p;
  };
  __bf16* roi_bf  = (__bf16*)alloc((size_t)NROI * 1024 * 2);
  __bf16* wq_bf   = (__bf16*)alloc((size_t)1024 * 1024 * 2);
  __bf16* wk_bf   = (__bf16*)alloc((size_t)1024 * 1024 * 2);
  __bf16* wc_bf   = (__bf16*)alloc((size_t)1024 * 1024 * 2);
  __bf16* wpos_bf = (__bf16*)alloc((size_t)16 * 64 * 2);
  __bf16* q_bf    = (__bf16*)alloc((size_t)NROI * 1024 * 2);
  __bf16* k_bf    = (__bf16*)alloc((size_t)NONGT * 1024 * 2);
  float*  waff    = (float*)alloc((size_t)GRP * NM * 4);     // 128 MB
  __bf16* p_bf    = (__bf16*)alloc((size_t)GRP * NM * 2);    // 64 MB
  __bf16* ot_bf   = (__bf16*)alloc((size_t)GRP * NM * 2);    // 64 MB

  auto cvt = [&](const float* src, __bf16* dst, long n) {
    cvt_kernel<<<(unsigned)((n + 255) / 256), 256, 0, stream>>>(src, dst, n);
  };
  cvt(roi,   roi_bf,  (long)NROI * 1024);
  cvt(Wq,    wq_bf,   1024L * 1024);
  cvt(Wk,    wk_bf,   1024L * 1024);
  cvt(Wconv, wc_bf,   1024L * 1024);
  cvt(Wpos,  wpos_bf, 16L * 64);

  // q = 0.125*(roi@Wq^T + bq)  (fold 1/sqrt(64)); k = nongt@Wk^T + bk
  gemm_nt_bf16<<<dim3(16, 32), 256, 0, stream>>>(roi_bf, wq_bf, bq, q_bf,
                                                 2048, 1024, 1024, 0.125f);
  gemm_nt_bf16<<<dim3(16, 16), 256, 0, stream>>>(roi_bf, wk_bf, bk, k_bf,
                                                 1024, 1024, 1024, 1.0f);
  // waff[g][n][m] = log-position term (streams the 536MB PE tensor once)
  posfeat_kernel<<<16384, 256, 0, stream>>>(pe, wpos_bf, bpos, waff);
  // waff += qb.kb^T
  affscale_kernel<<<dim3(32, 16, 16), 256, 0, stream>>>(q_bf, k_bf, waff);
  // softmax over m
  softmax_kernel<<<4096, 256, 0, stream>>>(waff, p_bf);
  // output_t[g] = P[g] @ nongt (nongt = first 1024 rows of roi_bf)
  outfeat_kernel<<<dim3(16, 16, 16), 256, 0, stream>>>(p_bf, roi_bf, ot_bf);
  // final grouped conv + bias -> d_out
  conv_kernel<<<dim3(1, 32, 16), 256, 0, stream>>>(ot_bf, wc_bf, bconv,
                                                   (float*)d_out);
}
